// PointcloudEncoder_88252987998635
// MI455X (gfx1250) — matrix operand order repaired and checked by reading, not compiled, not run on hardware
//
#include <hip/hip_runtime.h>
#include <hip/hip_bf16.h>

// ---------------------------------------------------------------------------
// Point cloud encoder for MI455X (gfx1250, wave32, WMMA + TDM).
// FPS -> KNN+group -> PointNet (TDM-staged WMMA f16 GEMMs, train-BN) ->
// e2t projection + pos-embed MLP -> assemble [B,513,768] f32.
// ---------------------------------------------------------------------------

#define BB 8
#define NN 16384
#define GG 512
#define KK 64
#define BG (BB * GG)        // 4096 groups
#define MM (BG * KK)        // 262144 points total
#define TRANS_ 768

typedef __attribute__((ext_vector_type(16))) _Float16 v16h;
typedef __attribute__((ext_vector_type(8)))  _Float16 v8h;
typedef __attribute__((ext_vector_type(8)))  float    v8f;
typedef __attribute__((ext_vector_type(4)))  unsigned int u32x4;
typedef __attribute__((ext_vector_type(8)))  int          i32x8;
typedef __attribute__((ext_vector_type(4)))  int          i32x4;

#if defined(__has_builtin)
#  if __has_builtin(__builtin_amdgcn_tensor_load_to_lds)
#    define USE_TDM 1
#  endif
#endif
#ifndef USE_TDM
#  define USE_TDM 0
#endif

// ---------------------------------------------------------------------------
// 1) Farthest point sampling: one block per batch, dist kept in registers.
// ---------------------------------------------------------------------------
__global__ __launch_bounds__(1024) void fps_kernel(
    const float* __restrict__ pts, float* __restrict__ center, int* __restrict__ fidx)
{
    const int b   = blockIdx.x;
    const int tid = threadIdx.x;
    const float* P = pts + (size_t)b * NN * 3;

    float d[16];
#pragma unroll
    for (int i = 0; i < 16; ++i) d[i] = 1e10f;

    __shared__ float rb[32];
    __shared__ int   ri[32];
    __shared__ int   sFar;
    __shared__ float sc[3];

    int far = 0;
    for (int it = 0; it < GG; ++it) {
        if (tid == 0) fidx[b * GG + it] = far;
        if (tid < 3) {
            float v = P[far * 3 + tid];
            center[((size_t)b * GG + it) * 3 + tid] = v;
            sc[tid] = v;
        }
        __syncthreads();
        const float cx = sc[0], cy = sc[1], cz = sc[2];

        float best = -1.0f; int bi = 0;
#pragma unroll
        for (int i = 0; i < 16; ++i) {
            const int n = tid + i * 1024;
            const float dx = P[n * 3 + 0] - cx;
            const float dy = P[n * 3 + 1] - cy;
            const float dz = P[n * 3 + 2] - cz;
            const float dd = dx * dx + dy * dy + dz * dz;
            d[i] = fminf(d[i], dd);
            if (d[i] > best) { best = d[i]; bi = n; }
        }
        for (int off = 16; off; off >>= 1) {               // wave32 argmax
            const float ob = __shfl_down(best, off);
            const int   oi = __shfl_down(bi, off);
            if (ob > best) { best = ob; bi = oi; }
        }
        if ((tid & 31) == 0) { rb[tid >> 5] = best; ri[tid >> 5] = bi; }
        __syncthreads();
        if (tid == 0) {
            for (int w = 1; w < 32; ++w)
                if (rb[w] > best) { best = rb[w]; bi = ri[w]; }
            sFar = bi;
        }
        __syncthreads();
        far = sFar;
    }
}

// ---------------------------------------------------------------------------
// 2) KNN (top-64 via iterative argmin over f16 LDS dists) + gather + feats.
//    feats: [MM, 32] f16; cols 0-2 rel-xyz, 3-5 color, 6-31 zero pad.
// ---------------------------------------------------------------------------
__global__ __launch_bounds__(256) void knn_feats_kernel(
    const float* __restrict__ pts, const float* __restrict__ colors,
    const float* __restrict__ center, _Float16* __restrict__ feats)
{
    const int bg  = blockIdx.x;
    const int b   = bg >> 9;
    const int tid = threadIdx.x;

    __shared__ _Float16 dist[NN];     // 32 KB
    __shared__ float rbest[8];
    __shared__ int   rbesti[8];
    __shared__ int   selIdx[KK];
    __shared__ float cc[3];

    if (tid < 3) cc[tid] = center[(size_t)bg * 3 + tid];
    __syncthreads();
    const float cx = cc[0], cy = cc[1], cz = cc[2];

    const float* P = pts    + (size_t)b * NN * 3;
    const float* C = colors + (size_t)b * NN * 3;

    for (int n = tid; n < NN; n += 256) {
        const float dx = P[n * 3 + 0] - cx;
        const float dy = P[n * 3 + 1] - cy;
        const float dz = P[n * 3 + 2] - cz;
        dist[n] = (_Float16)(dx * dx + dy * dy + dz * dz);
    }
    __syncthreads();

    for (int j = 0; j < KK; ++j) {
        float best = 1e30f; int bi = 0;
        for (int n = tid; n < NN; n += 256) {
            const float dd = (float)dist[n];
            if (dd < best) { best = dd; bi = n; }
        }
        for (int off = 16; off; off >>= 1) {
            const float ob = __shfl_down(best, off);
            const int   oi = __shfl_down(bi, off);
            if (ob < best) { best = ob; bi = oi; }
        }
        if ((tid & 31) == 0) { rbest[tid >> 5] = best; rbesti[tid >> 5] = bi; }
        __syncthreads();
        if (tid == 0) {
            for (int w = 1; w < 8; ++w)
                if (rbest[w] < best) { best = rbest[w]; bi = rbesti[w]; }
            selIdx[j] = bi;
            dist[bi] = (_Float16)65504.0f;
        }
        __syncthreads();
    }

    for (int j = tid; j < KK; j += 256) {
        const int n = selIdx[j];
        _Float16* f = feats + ((size_t)bg * KK + j) * 32;
        f[0] = (_Float16)(P[n * 3 + 0] - cx);
        f[1] = (_Float16)(P[n * 3 + 1] - cy);
        f[2] = (_Float16)(P[n * 3 + 2] - cz);
        f[3] = (_Float16)C[n * 3 + 0];
        f[4] = (_Float16)C[n * 3 + 1];
        f[5] = (_Float16)C[n * 3 + 2];
#pragma unroll
        for (int q = 6; q < 32; ++q) f[q] = (_Float16)0.0f;
    }
}

// ---------------------------------------------------------------------------
// 3) Pack f32 weights W[N, KcSrc] (cols koff..koff+Kval) into wave32 WMMA
//    B-fragment layout: Bp[((kt*Ntiles+nt)*32+lane)*16+e],
//    k = kt*32 + (e>=8?16:0) + (lane>=16?8:0) + (e&7), n = nt*16 + (lane&15).   (ISA 7.12.2)
// ---------------------------------------------------------------------------
__global__ void pack_w_kernel(const float* __restrict__ W, _Float16* __restrict__ Bp,
                              int KcSrc, int koff, int Kval, int Kpad, int Nn)
{
    const int idx   = blockIdx.x * 256 + threadIdx.x;
    const int total = (Kpad >> 5) * (Nn >> 4) * 512;
    if (idx >= total) return;
    const int e      = idx & 15;
    const int lane   = (idx >> 4) & 31;
    const int t      = idx >> 9;
    const int ntiles = Nn >> 4;
    const int nt     = t % ntiles;
    const int kt     = t / ntiles;
    const int k      = kt * 32 + ((e >> 3) << 4) + ((lane >> 4) << 3) + (e & 7);
    const int n      = nt * 16 + (lane & 15);
    _Float16 v = (_Float16)0.0f;
    if (k < Kval) v = (_Float16)W[(size_t)n * KcSrc + koff + k];
    Bp[idx] = v;
}

// ---------------------------------------------------------------------------
// 4) TDM issue helper: 2-D tile (rows x cols of f16) global -> LDS.
//    D# layout per CDNA5 ISA 8.3/8.4; count=1, type=2, data_size=1 (2 bytes).
// ---------------------------------------------------------------------------
#if USE_TDM
__device__ __forceinline__ void tdm_load_tile(
    unsigned lds_addr, const void* gsrc,
    unsigned tensor_d0, unsigned tensor_d1, unsigned long long stride0,
    unsigned tile_d0, unsigned tile_d1)
{
    const unsigned long long ga = (unsigned long long)(uintptr_t)gsrc;
    u32x4 g0;
    g0.x = 1u;                                              // count=1 (valid)
    g0.y = lds_addr;                                        // LDS byte address
    g0.z = (unsigned)ga;                                    // global_addr lo
    g0.w = (unsigned)((ga >> 32) & 0x01FFFFFFu) | 0x80000000u; // hi | type=2
    i32x8 g1;
    g1[0] = (int)(1u << 16);                                // data_size = 2B
    g1[1] = (int)((tensor_d0 & 0xFFFFu) << 16);             // td0[15:0]
    g1[2] = (int)(((tensor_d0 >> 16) & 0xFFFFu) | ((tensor_d1 & 0xFFFFu) << 16));
    g1[3] = (int)(((tensor_d1 >> 16) & 0xFFFFu) | ((tile_d0 & 0xFFFFu) << 16));
    g1[4] = (int)(tile_d1 & 0xFFFFu);                       // tile_dim2 = 0
    g1[5] = (int)(unsigned)(stride0 & 0xFFFFFFFFull);
    g1[6] = (int)(unsigned)((stride0 >> 32) & 0xFFFFull);
    g1[7] = 0;
    i32x4 z4 = {};
#if __clang_major__ >= 23
    i32x8 z8 = {};
    __builtin_amdgcn_tensor_load_to_lds(g0, g1, z4, z4, z8, 0);
#else
    __builtin_amdgcn_tensor_load_to_lds(g0, g1, z4, z4, 0);
#endif
}
#endif

// ---------------------------------------------------------------------------
// 5) TDM-staged, double-buffered WMMA GEMM.
//    Block = 256 thr = 8 waves; block tile 128(M) x 64(N); each wave owns one
//    16-row M-tile and computes 4 N-tiles per A fragment (4 WMMAs / A load).
//    PRO: 0 = plain f16 A; 1 = BN(mean,rstd,gamma,beta)+ReLU on A
//    EPI: 0 = f16 + bias; 1 = f32 + bias; 2 = atomic group-max (f32 [BG,N]);
//         3 = f16 + per-group row addend (grow), no bias
// ---------------------------------------------------------------------------
__device__ __forceinline__ void atomic_max_f32(float* addr, float v)
{
    if (v >= 0.0f) atomicMax((int*)addr, __float_as_int(v));
    else           atomicMin((unsigned int*)addr, __float_as_uint(v));
}

template <int PRO, int EPI>
__global__ __launch_bounds__(256) void wmma_gemm_kernel(
    const _Float16* __restrict__ A, int lda, int Mrows,
    const _Float16* __restrict__ Bp,
    const float* __restrict__ bias,
    const float* __restrict__ mean, const float* __restrict__ rstd,
    const float* __restrict__ gam,  const float* __restrict__ bet,
    const float* __restrict__ grow,
    _Float16* __restrict__ Ch, float* __restrict__ Cf, int ldc,
    int Kpad, int Ntiles)
{
    __shared__ _Float16 sA[2][128 * 32];   // 8 KB x2
    __shared__ _Float16 sB[2][4 * 512];    // 4 KB x2

    const int lane    = threadIdx.x & 31;
    const int wid     = threadIdx.x >> 5;
    const int m0      = blockIdx.y * 128;
    const int ntile0  = blockIdx.x * 4;
    const int laneoff = (lane >> 4) << 3;                 // 0 or 8
    const int mrow    = wid * 16 + (lane & 15);           // A row in block tile
    const int ktiles  = Kpad >> 5;

    auto stage = [&](int kt, int buf) {
#if USE_TDM
        if (wid == 0) {
            // A panel: rows m0..m0+127, cols kt*32..kt*32+31
            tdm_load_tile((unsigned)(uintptr_t)&sA[buf][0],
                          A + (size_t)m0 * lda + (kt << 5),
                          (unsigned)lda, (unsigned)(Mrows - m0),
                          (unsigned long long)lda, 32u, 128u);
            // B panel: 4 consecutive n-tiles, contiguous 2048 halfs
            tdm_load_tile((unsigned)(uintptr_t)&sB[buf][0],
                          Bp + ((size_t)kt * Ntiles + ntile0) * 512,
                          2048u, 1u, 2048ull, 2048u, 1u);
        }
#else
        // cooperative fallback staging (all 256 threads)
        const int t   = threadIdx.x;
        const int row = t >> 1, h16 = (t & 1) << 4;       // 128 rows x 2 chunks
        *(v16h*)&sA[buf][row * 32 + h16] =
            *(const v16h*)(A + (size_t)(m0 + row) * lda + (kt << 5) + h16);
        *(v8h*)&sB[buf][t * 8] =
            *(const v8h*)(Bp + ((size_t)kt * Ntiles + ntile0) * 512 + t * 8);
#endif
    };

    stage(0, 0);

    v8f acc[4] = {};
    for (int kt = 0; kt < ktiles; ++kt) {
        const int cur = kt & 1;
#if USE_TDM
        if (wid == 0) __builtin_amdgcn_s_wait_tensorcnt(0);
#endif
        __syncthreads();
        if (kt + 1 < ktiles) stage(kt + 1, cur ^ 1);

        // ---- A fragment from LDS (ds_load_b128 x2) + prologue transform ----
        const _Float16* ap = &sA[cur][mrow * 32];
        const v8h lo = *(const v8h*)(ap + laneoff);
        const v8h hi = *(const v8h*)(ap + 16 + laneoff);
        v16h a;
        if constexpr (PRO == 1) {
            const int kb = kt << 5;
#pragma unroll
            for (int e = 0; e < 8; ++e) {
                const int c0 = kb + laneoff + e;
                float x = (float)lo[e];
                x = fmaxf((x - mean[c0]) * rstd[c0] * gam[c0] + bet[c0], 0.0f);
                a[e] = (_Float16)x;
                const int c1 = kb + 16 + laneoff + e;
                float y = (float)hi[e];
                y = fmaxf((y - mean[c1]) * rstd[c1] * gam[c1] + bet[c1], 0.0f);
                a[8 + e] = (_Float16)y;
            }
        } else {
#pragma unroll
            for (int e = 0; e < 8; ++e) { a[e] = lo[e]; a[8 + e] = hi[e]; }
        }

        // ---- 4 B fragments + 4 WMMAs ----
        const _Float16* bp = &sB[cur][lane * 16];
        const v16h b0 = *(const v16h*)(bp);
        const v16h b1 = *(const v16h*)(bp + 512);
        const v16h b2 = *(const v16h*)(bp + 1024);
        const v16h b3 = *(const v16h*)(bp + 1536);
        acc[0] = __builtin_amdgcn_wmma_f32_16x16x32_f16(false, a, false, b0, (short)0, acc[0], false, false);
        acc[1] = __builtin_amdgcn_wmma_f32_16x16x32_f16(false, a, false, b1, (short)0, acc[1], false, false);
        acc[2] = __builtin_amdgcn_wmma_f32_16x16x32_f16(false, a, false, b2, (short)0, acc[2], false, false);
        acc[3] = __builtin_amdgcn_wmma_f32_16x16x32_f16(false, a, false, b3, (short)0, acc[3], false, false);
    }

    // ---- epilogue ----
    const int mbase = m0 + wid * 16 + ((lane >> 4) << 3);
#pragma unroll
    for (int j = 0; j < 4; ++j) {
        const int n = (ntile0 + j) * 16 + (lane & 15);
        float bn = 0.0f;
        if constexpr (EPI != 3) bn = bias[n];
#pragma unroll
        for (int r = 0; r < 8; ++r) {
            const int m = mbase + r;
            float v = acc[j][r] + bn;
            if constexpr (EPI == 0) {
                Ch[(size_t)m * ldc + n] = (_Float16)v;
            } else if constexpr (EPI == 1) {
                Cf[(size_t)m * ldc + n] = v;
            } else if constexpr (EPI == 2) {
                atomic_max_f32(Cf + (size_t)(m >> 6) * ldc + n, v);
            } else {
                v += grow[(size_t)(m >> 6) * ldc + n];
                Ch[(size_t)m * ldc + n] = (_Float16)v;
            }
        }
    }
}

// ---------------------------------------------------------------------------
// 6) Train-mode BatchNorm stats: one block per channel, reduce over M rows.
// ---------------------------------------------------------------------------
__global__ __launch_bounds__(256) void bn_stats_kernel(
    const _Float16* __restrict__ H, int Mrows, int C,
    float* __restrict__ mean, float* __restrict__ rstd)
{
    const int c   = blockIdx.x;
    const int tid = threadIdx.x;
    float s = 0.0f, s2 = 0.0f;
    for (int m = tid; m < Mrows; m += 256) {
        __builtin_prefetch(H + (size_t)(m + 2048) * C + c, 0, 1);
        const float x = (float)H[(size_t)m * C + c];
        s += x; s2 += x * x;
    }
    __shared__ float ss[256], sq[256];
    ss[tid] = s; sq[tid] = s2;
    __syncthreads();
    for (int o = 128; o; o >>= 1) {
        if (tid < o) { ss[tid] += ss[tid + o]; sq[tid] += sq[tid + o]; }
        __syncthreads();
    }
    if (tid == 0) {
        const float mu  = ss[0] / (float)Mrows;
        const float var = sq[0] / (float)Mrows - mu * mu;
        mean[c] = mu;
        rstd[c] = rsqrtf(var + 1e-5f);
    }
}

__global__ void group_max_kernel(const _Float16* __restrict__ H,
                                 _Float16* __restrict__ Gm, int C)
{
    const int idx = blockIdx.x * 256 + threadIdx.x;
    if (idx >= BG * C) return;
    const int gg = idx / C;
    const int c  = idx % C;
    const _Float16* p = H + ((size_t)gg * KK) * C + c;
    float m = -1e30f;
#pragma unroll 8
    for (int i = 0; i < KK; ++i) m = fmaxf(m, (float)p[(size_t)i * C]);
    Gm[(size_t)gg * C + c] = (_Float16)m;
}

__global__ void fill_neg_kernel(float* __restrict__ p, int n)
{
    const int i = blockIdx.x * 256 + threadIdx.x;
    if (i < n) p[i] = -3.0e38f;
}

__global__ void f32_to_f16_kernel(const float* __restrict__ s,
                                  _Float16* __restrict__ d, int n)
{
    const int i = blockIdx.x * 256 + threadIdx.x;
    if (i < n) d[i] = (_Float16)s[i];
}

// pos_embed layer 1: center[4096,3] @ wp1^T + bp1 -> exact GELU -> f16
__global__ void pos_mlp1_kernel(const float* __restrict__ center,
                                const float* __restrict__ wp1,
                                const float* __restrict__ bp1,
                                _Float16* __restrict__ posh)
{
    const int idx = blockIdx.x * 256 + threadIdx.x;
    if (idx >= BG * 128) return;
    const int r = idx >> 7;
    const int c = idx & 127;
    const float* ct = center + (size_t)r * 3;
    float h = bp1[c] + ct[0] * wp1[c * 3 + 0]
                     + ct[1] * wp1[c * 3 + 1]
                     + ct[2] * wp1[c * 3 + 2];
    h = 0.5f * h * (1.0f + erff(h * 0.70710678118654752f));
    posh[idx] = (_Float16)h;
}

__global__ void assemble_kernel(const float* __restrict__ tokp,
                                const float* __restrict__ pos2,
                                const float* __restrict__ clsT,
                                const float* __restrict__ clsP,
                                float* __restrict__ out)
{
    const int idx = blockIdx.x * 256 + threadIdx.x;
    const int total = BB * (GG + 1) * TRANS_;
    if (idx >= total) return;
    const int c = idx % TRANS_;
    const int t = (idx / TRANS_) % (GG + 1);
    const int b = idx / (TRANS_ * (GG + 1));
    float v;
    if (t == 0) {
        v = clsT[c] + clsP[c];
    } else {
        const size_t r = (size_t)b * GG + (t - 1);
        v = tokp[r * TRANS_ + c] + pos2[r * TRANS_ + c];
    }
    out[idx] = v;
}

// ---------------------------------------------------------------------------
// Host launcher
// ---------------------------------------------------------------------------
extern "C" void kernel_launch(void* const* d_in, const int* in_sizes, int n_in,
                              void* d_out, int out_size, void* d_ws, size_t ws_size,
                              hipStream_t stream)
{
    (void)in_sizes; (void)n_in; (void)out_size; (void)ws_size;

    const float* pts    = (const float*)d_in[0];
    const float* colors = (const float*)d_in[1];
    const float* w1     = (const float*)d_in[2];
    const float* b1     = (const float*)d_in[3];
    const float* g1     = (const float*)d_in[4];
    const float* be1    = (const float*)d_in[5];
    const float* w2     = (const float*)d_in[6];
    const float* b2     = (const float*)d_in[7];
    const float* w3     = (const float*)d_in[8];
    const float* b3     = (const float*)d_in[9];
    const float* g2     = (const float*)d_in[10];
    const float* be2    = (const float*)d_in[11];
    const float* w4     = (const float*)d_in[12];
    const float* b4     = (const float*)d_in[13];
    const float* w_e2t  = (const float*)d_in[14];
    const float* b_e2t  = (const float*)d_in[15];
    const float* clsT   = (const float*)d_in[16];
    const float* clsP   = (const float*)d_in[17];
    const float* wp1    = (const float*)d_in[18];
    const float* bp1    = (const float*)d_in[19];
    const float* wp2    = (const float*)d_in[20];
    const float* bp2    = (const float*)d_in[21];
    float* out = (float*)d_out;

    char* base = (char*)d_ws;
    size_t off = 0;
    auto take = [&](size_t bytes) -> void* {
        void* p = base + off;
        off += (bytes + 255) & ~(size_t)255;
        return p;
    };
    float*    center = (float*)   take((size_t)BG * 3 * 4);
    int*      fidx   = (int*)     take((size_t)BB * GG * 4);
    _Float16* feats  = (_Float16*)take((size_t)MM * 32 * 2);
    _Float16* h1     = (_Float16*)take((size_t)MM * 128 * 2);
    _Float16* h2     = (_Float16*)take((size_t)MM * 256 * 2);
    _Float16* gmax   = (_Float16*)take((size_t)BG * 256 * 2);
    _Float16* h3     = (_Float16*)take((size_t)MM * 512 * 2);
    float*    Cg     = (float*)   take((size_t)BG * 512 * 4);
    float*    tokf   = (float*)   take((size_t)BG * 512 * 4);
    _Float16* tok16  = (_Float16*)take((size_t)BG * 512 * 2);
    _Float16* posh   = (_Float16*)take((size_t)BG * 128 * 2);
    float*    tokp   = (float*)   take((size_t)BG * TRANS_ * 4);
    float*    pos2   = (float*)   take((size_t)BG * TRANS_ * 4);
    float*    mean1  = (float*)   take(128 * 4);
    float*    rstd1  = (float*)   take(128 * 4);
    float*    mean2  = (float*)   take(512 * 4);
    float*    rstd2  = (float*)   take(512 * 4);
    _Float16* W1p    = (_Float16*)take((size_t)1  * 8  * 512 * 2);
    _Float16* W2p    = (_Float16*)take((size_t)4  * 16 * 512 * 2);
    _Float16* W3ap   = (_Float16*)take((size_t)8  * 32 * 512 * 2);
    _Float16* W3bp   = (_Float16*)take((size_t)8  * 32 * 512 * 2);
    _Float16* W4p    = (_Float16*)take((size_t)16 * 32 * 512 * 2);
    _Float16* Wep    = (_Float16*)take((size_t)16 * 48 * 512 * 2);
    _Float16* Wp2p   = (_Float16*)take((size_t)4  * 48 * 512 * 2);

    // ---- geometry ----
    fps_kernel<<<BB, 1024, 0, stream>>>(pts, center, fidx);
    knn_feats_kernel<<<BG, 256, 0, stream>>>(pts, colors, center, feats);

    // ---- weight packing ----
    auto pack = [&](const float* W, _Float16* Bp, int KcSrc, int koff,
                    int Kval, int Kpad, int Nn) {
        const int total = (Kpad / 32) * (Nn / 16) * 512;
        pack_w_kernel<<<(total + 255) / 256, 256, 0, stream>>>(
            W, Bp, KcSrc, koff, Kval, Kpad, Nn);
    };
    pack(w1,    W1p,  6,   0,   6,   32,  128);
    pack(w2,    W2p,  128, 0,   128, 128, 256);
    pack(w3,    W3ap, 512, 0,   256, 256, 512);   // gmax half of W3
    pack(w3,    W3bp, 512, 256, 256, 256, 512);   // h2 half of W3
    pack(w4,    W4p,  512, 0,   512, 512, 512);
    pack(w_e2t, Wep,  512, 0,   512, 512, TRANS_);
    pack(wp2,   Wp2p, 128, 0,   128, 128, TRANS_);

    // ---- PointNet encoder (TDM-staged WMMA) ----
    // L1: feats[MM,32] @ W1 -> h1 f16 [MM,128]
    wmma_gemm_kernel<0, 0><<<dim3(2, MM / 128), 256, 0, stream>>>(
        feats, 32, MM, W1p, b1, nullptr, nullptr, nullptr, nullptr, nullptr,
        h1, nullptr, 128, 32, 8);
    bn_stats_kernel<<<128, 256, 0, stream>>>(h1, MM, 128, mean1, rstd1);
    // L2: relu(bn(h1)) @ W2 -> h2 f16 [MM,256]
    wmma_gemm_kernel<1, 0><<<dim3(4, MM / 128), 256, 0, stream>>>(
        h1, 128, MM, W2p, b2, mean1, rstd1, g1, be1, nullptr,
        h2, nullptr, 256, 128, 16);
    group_max_kernel<<<(BG * 256 + 255) / 256, 256, 0, stream>>>(h2, gmax, 256);
    // L3a: per-group gmax @ W3[:, :256] + b3 -> Cg f32 [BG,512]
    wmma_gemm_kernel<0, 1><<<dim3(8, BG / 128), 256, 0, stream>>>(
        gmax, 256, BG, W3ap, b3, nullptr, nullptr, nullptr, nullptr, nullptr,
        nullptr, Cg, 512, 256, 32);
    // L3b: h2 @ W3[:, 256:] + Cg[group] -> h3 f16 [MM,512]
    wmma_gemm_kernel<0, 3><<<dim3(8, MM / 128), 256, 0, stream>>>(
        h2, 256, MM, W3bp, b3, nullptr, nullptr, nullptr, nullptr, Cg,
        h3, nullptr, 512, 256, 32);
    bn_stats_kernel<<<512, 256, 0, stream>>>(h3, MM, 512, mean2, rstd2);
    // L4: relu(bn(h3)) @ W4 + b4 -> fused group-max into tokf f32 [BG,512]
    fill_neg_kernel<<<(BG * 512 + 255) / 256, 256, 0, stream>>>(tokf, BG * 512);
    wmma_gemm_kernel<1, 2><<<dim3(8, MM / 128), 256, 0, stream>>>(
        h3, 512, MM, W4p, b4, mean2, rstd2, g2, be2, nullptr,
        nullptr, tokf, 512, 512, 32);
    f32_to_f16_kernel<<<(BG * 512 + 255) / 256, 256, 0, stream>>>(tokf, tok16, BG * 512);

    // ---- projections ----
    wmma_gemm_kernel<0, 1><<<dim3(12, BG / 128), 256, 0, stream>>>(
        tok16, 512, BG, Wep, b_e2t, nullptr, nullptr, nullptr, nullptr, nullptr,
        nullptr, tokp, TRANS_, 512, 48);
    pos_mlp1_kernel<<<(BG * 128 + 255) / 256, 256, 0, stream>>>(center, wp1, bp1, posh);
    wmma_gemm_kernel<0, 1><<<dim3(12, BG / 128), 256, 0, stream>>>(
        posh, 128, BG, Wp2p, bp2, nullptr, nullptr, nullptr, nullptr, nullptr,
        nullptr, pos2, TRANS_, 128, 48);

    // ---- final ----
    const int total = BB * (GG + 1) * TRANS_;
    assemble_kernel<<<(total + 255) / 256, 256, 0, stream>>>(tokp, pos2, clsT, clsP, out);
}
